// VectorQuantizer_2456721293856
// MI455X (gfx1250) — matrix operand order repaired and checked by reading, not compiled, not run on hardware
//
#include <hip/hip_runtime.h>

typedef __attribute__((ext_vector_type(16))) __bf16 bf16x16;
typedef __attribute__((ext_vector_type(8)))  __bf16 bf16x8;
typedef __attribute__((ext_vector_type(8)))  float  f32x8;

#define N_ROWS          65536
#define DIM             256
#define K_CB            1024
#define ROWS_PER_BLOCK  128
#define COLS_PER_STAGE  64
#define STAGES          (K_CB / COLS_PER_STAGE)          // 16
#define LDS_STRIDE      264   // 256 bf16 + 8 pad -> 4-bank skew, conflict-free b128 reads
#define BUF_ELEMS       (COLS_PER_STAGE * LDS_STRIDE)    // 16896 bf16
#define BUF_BYTES       (BUF_ELEMS * 2)                  // 33792 B per buffer

// ---------------- CDNA5 async global->LDS helpers ----------------
__device__ __forceinline__ void async_b128(unsigned ldsOff, const void* gaddr) {
  // GLOBAL_LOAD_ASYNC_TO_LDS_B128: VDST = LDS byte address (VGPR),
  // VADDR = 64-bit global address (VGPR pair), SADDR = off. Tracked by ASYNCcnt.
  asm volatile("global_load_async_to_lds_b128 %0, %1, off"
               :
               : "v"(ldsOff), "v"((unsigned long long)(uintptr_t)gaddr)
               : "memory");
}
__device__ __forceinline__ void wait_async0() {
  asm volatile("s_wait_asynccnt 0x0" ::: "memory");
}

// ---------------- kernel 1: E -> bf16, ||e||^2 ----------------
__global__ void vq_prep_embed(const float* __restrict__ E,
                              __bf16* __restrict__ ebf,
                              float* __restrict__ enorm) {
  int r = blockIdx.x;
  int t = threadIdx.x;                 // 64 threads
  float4 v = reinterpret_cast<const float4*>(E + (size_t)r * DIM)[t];
  __bf16* dst = ebf + (size_t)r * DIM + t * 4;
  dst[0] = (__bf16)v.x; dst[1] = (__bf16)v.y;
  dst[2] = (__bf16)v.z; dst[3] = (__bf16)v.w;
  __shared__ float red[64];
  red[t] = v.x * v.x + v.y * v.y + v.z * v.z + v.w * v.w;
  __syncthreads();
  if (t == 0) {
    float s = 0.f;
    for (int i = 0; i < 64; ++i) s += red[i];
    enorm[r] = s;
  }
}

// ---------------- kernel 2: WMMA distance GEMM + argmin ----------------
__device__ __forceinline__ void issue_stage(const __bf16* __restrict__ ebf,
                                            int stageBase, unsigned bufOff, int tid) {
  // Stage 64 codebook rows x 256 bf16 = 32 KB into LDS (padded), async DMA.
#pragma unroll
  for (int i = 0; i < 8; ++i) {
    int u   = i * 256 + tid;   // 16-byte units: 64 cols * 32 units
    int cc  = u >> 5;          // col within stage
    int o16 = u & 31;          // 16B unit within the 512B row
    const __bf16* g = ebf + (size_t)(stageBase + cc) * DIM + o16 * 8;
    unsigned l = bufOff + (unsigned)(cc * (LDS_STRIDE * 2) + o16 * 16);
    async_b128(l, g);
  }
}

__device__ __forceinline__ bf16x16 loadB(const __bf16* base, int c) {
  const __bf16* bp = base + c * 32;
  bf16x8 b0 = *reinterpret_cast<const bf16x8*>(bp);
  bf16x8 b1 = *reinterpret_cast<const bf16x8*>(bp + 16);
  return __builtin_shufflevector(b0, b1, 0,1,2,3,4,5,6,7,8,9,10,11,12,13,14,15);
}

__global__ void __launch_bounds__(256)
vq_argmin_wmma(const float* __restrict__ X,
               const __bf16* __restrict__ ebf,
               const float* __restrict__ enorm,
               int* __restrict__ outIdx) {
  extern __shared__ __bf16 sB[];       // 2 * BUF_ELEMS (double buffer)
  const int tid  = threadIdx.x;
  const int lane = tid & 31;
  const int wave = tid >> 5;
  const int h    = lane >> 4;          // half-wave
  const int n16  = lane & 15;
  const int rowBase = blockIdx.x * ROWS_PER_BLOCK + wave * 16;

  // Kick off stage 0 DMA before doing register work.
  issue_stage(ebf, 0, 0u, tid);

  // Preload A fragments: 16 rows x 256 (bf16), ISA 16-bit A 16x32 layout:
  // elem e<8 -> K = h*8 + e ; elem e>=8 -> K = 16 + h*8 + (e-8)
  bf16x16 afrag[8];
  {
    const float4* xr = reinterpret_cast<const float4*>(X + (size_t)(rowBase + n16) * DIM);
#pragma unroll
    for (int c = 0; c < 8; ++c) {
      float4 l0 = xr[c * 8 + h * 2 + 0];
      float4 l1 = xr[c * 8 + h * 2 + 1];
      float4 h0 = xr[c * 8 + 4 + h * 2 + 0];
      float4 h1 = xr[c * 8 + 4 + h * 2 + 1];
      bf16x16 a;
      a[0]=(__bf16)l0.x;  a[1]=(__bf16)l0.y;  a[2]=(__bf16)l0.z;  a[3]=(__bf16)l0.w;
      a[4]=(__bf16)l1.x;  a[5]=(__bf16)l1.y;  a[6]=(__bf16)l1.z;  a[7]=(__bf16)l1.w;
      a[8]=(__bf16)h0.x;  a[9]=(__bf16)h0.y;  a[10]=(__bf16)h0.z; a[11]=(__bf16)h0.w;
      a[12]=(__bf16)h1.x; a[13]=(__bf16)h1.y; a[14]=(__bf16)h1.z; a[15]=(__bf16)h1.w;
      afrag[c] = a;
    }
  }

  float bestVal[8];
  int   bestIdx[8];
#pragma unroll
  for (int v = 0; v < 8; ++v) { bestVal[v] = 3.4e38f; bestIdx[v] = 0; }

  for (int s = 0; s < STAGES; ++s) {
    const int stageBase = s * COLS_PER_STAGE;
    wait_async0();           // our async writes for stage s complete
    __syncthreads();         // everyone's writes visible; prev buffer fully read
    if (s + 1 < STAGES)      // overlap next stage's DMA with this stage's WMMAs
      issue_stage(ebf, stageBase + COLS_PER_STAGE,
                  (unsigned)(((s + 1) & 1) * BUF_BYTES), tid);

    const __bf16* buf = sB + (s & 1) * BUF_ELEMS;
#pragma unroll
    for (int ct = 0; ct < COLS_PER_STAGE / 16; ++ct) {
      const __bf16* base = buf + (ct * 16 + n16) * LDS_STRIDE + h * 8;
      f32x8 acc = {};
      bf16x16 bc = loadB(base, 0);             // software-pipelined B fragments
#pragma unroll
      for (int c = 0; c < 8; ++c) {
        bf16x16 bn = (c < 7) ? loadB(base, c + 1) : bc;
        acc = __builtin_amdgcn_wmma_f32_16x16x32_bf16(
            false, afrag[c], false, bc, (short)0, acc, false, false);
        bc = bn;
      }
      const int col = stageBase + ct * 16 + n16;
      const float en = enorm[col];
#pragma unroll
      for (int v = 0; v < 8; ++v) {            // argmin key: ||e||^2 - 2 x.e
        float dist = en - 2.f * acc[v];
        if (dist < bestVal[v]) { bestVal[v] = dist; bestIdx[v] = col; }
      }
    }
  }

  // cross-lane argmin within each 16-lane half (row m = v + h*8)
#pragma unroll
  for (int v = 0; v < 8; ++v) {
    float bv = bestVal[v];
    int   bi = bestIdx[v];
#pragma unroll
    for (int mask = 1; mask <= 8; mask <<= 1) {
      float ov = __shfl_xor(bv, mask, 32);
      int   oi = __shfl_xor(bi, mask, 32);
      if (ov < bv || (ov == bv && oi < bi)) { bv = ov; bi = oi; }
    }
    if (n16 == 0) outIdx[rowBase + h * 8 + v] = bi;
  }
}

// ---------------- kernel 3: gather + write quantized + loss partials ----------------
__global__ void __launch_bounds__(256)
vq_gather_loss(const float* __restrict__ X, const float* __restrict__ E,
               const int* __restrict__ idx, float* __restrict__ out,
               float* __restrict__ partials) {
  int row = blockIdx.x * 4 + (threadIdx.x >> 6);
  int q   = threadIdx.x & 63;
  int k   = idx[row];
  float4 e = reinterpret_cast<const float4*>(E + (size_t)k * DIM)[q];
  float4 x = reinterpret_cast<const float4*>(X + (size_t)row * DIM)[q];
  reinterpret_cast<float4*>(out + (size_t)row * DIM)[q] = e;
  float dx = e.x - x.x, dy = e.y - x.y, dz = e.z - x.z, dw = e.w - x.w;
  __shared__ float red[256];
  red[threadIdx.x] = dx * dx + dy * dy + dz * dz + dw * dw;
  __syncthreads();
  for (int s = 128; s > 0; s >>= 1) {
    if (threadIdx.x < s) red[threadIdx.x] += red[threadIdx.x + s];
    __syncthreads();
  }
  if (threadIdx.x == 0) partials[blockIdx.x] = red[0];
}

// ---------------- kernel 4: final loss reduction ----------------
__global__ void vq_loss_final(const float* __restrict__ partials,
                              float* __restrict__ lossOut) {
  __shared__ float red[256];
  float s = 0.f;
  for (int i = threadIdx.x; i < N_ROWS / 4; i += 256) s += partials[i];
  red[threadIdx.x] = s;
  __syncthreads();
  for (int st = 128; st > 0; st >>= 1) {
    if (threadIdx.x < st) red[threadIdx.x] += red[threadIdx.x + st];
    __syncthreads();
  }
  if (threadIdx.x == 0)
    *lossOut = 1.25f * red[0] / (float)(N_ROWS * DIM);  // q_loss + 0.25*e_loss
}

extern "C" void kernel_launch(void* const* d_in, const int* in_sizes, int n_in,
                              void* d_out, int out_size, void* d_ws, size_t ws_size,
                              hipStream_t stream) {
  (void)in_sizes; (void)n_in; (void)out_size; (void)ws_size;
  const float* X = (const float*)d_in[0];   // [65536, 256]
  const float* E = (const float*)d_in[1];   // [1024, 256]
  float* out = (float*)d_out;               // [65536*256] quantized + [1] loss

  char* ws = (char*)d_ws;
  __bf16* ebf   = (__bf16*)(ws);            // 1024*256*2   = 524288 B
  float*  enorm = (float*)(ws + 524288);    // 1024*4       = 4096 B
  int*    idx   = (int*)  (ws + 528384);    // 65536*4      = 262144 B
  float*  part  = (float*)(ws + 790528);    // 16384*4      = 65536 B

  vq_prep_embed<<<K_CB, 64, 0, stream>>>(E, ebf, enorm);
  vq_argmin_wmma<<<N_ROWS / ROWS_PER_BLOCK, 256, 2 * BUF_BYTES, stream>>>(
      X, ebf, enorm, idx);
  vq_gather_loss<<<N_ROWS / 4, 256, 0, stream>>>(X, E, idx, out, part);
  vq_loss_final<<<1, 256, 0, stream>>>(part, out + (size_t)N_ROWS * DIM);
}
